// CausalSelfAttention_57191784513708
// MI455X (gfx1250) — compile-verified, matrix operand order checked by reading, not run on hardware
//
#include <hip/hip_runtime.h>
#include <stdint.h>

// ---------------------------------------------------------------------------
// Types for CDNA5 WMMA (wave32): V_WMMA_F32_16X16X32_BF16
// ---------------------------------------------------------------------------
typedef __bf16 bf16;
typedef __attribute__((ext_vector_type(16))) __bf16 v16bf;
typedef __attribute__((ext_vector_type(8)))  __bf16 v8bf;
typedef __attribute__((ext_vector_type(4)))  __bf16 v4bf;
typedef __attribute__((ext_vector_type(8)))  float  v8f;

#define WMMA_BF16(A, B, C) \
  __builtin_amdgcn_wmma_f32_16x16x32_bf16(false, (A), false, (B), (short)0, (C), false, false)

// Problem constants
#define BB    2
#define TT    2048
#define CC    1024
#define C3    3072
#define NHEAD 16
#define HD    64

__device__ __forceinline__ bf16 f2bf(float f) {
  union { float f; uint32_t u; } v; v.f = f;
  uint32_t r = v.u + 0x7fffu + ((v.u >> 16) & 1u);   // round-to-nearest-even
  union { uint16_t s; bf16 b; } o; o.s = (uint16_t)(r >> 16);
  return o.b;
}

// Load one 16-bf16 fragment: two contiguous 16-byte chunks at p and p+16 elems.
// Matches the ISA 16-bit A/B layout: lanes 0-15 hold K {0..7, 16..23} (lh=0),
// lanes 16-31 hold K {8..15, 24..31} (lh=8); caller bakes lh into p.
__device__ __forceinline__ v16bf load_frag16(const bf16* p) {
  v8bf lo = *(const v8bf*)p;
  v8bf hi = *(const v8bf*)(p + 16);
  v16bf r;
#pragma unroll
  for (int i = 0; i < 8; ++i) { r[i] = lo[i]; r[i + 8] = hi[i]; }
  return r;
}

// Raw LDS byte-offset of a pointer into a __shared__ object (for async-to-LDS).
__device__ __forceinline__ uint32_t lds_off(const void* p) {
  return (uint32_t)(uintptr_t)(__attribute__((address_space(3))) const void*)p;
}

// Async DMA: 16 bytes global -> LDS per lane, tracked by ASYNCcnt.
__device__ __forceinline__ void async_copy_b128(uint32_t lds_byte_off, const void* gptr) {
  asm volatile("global_load_async_to_lds_b128 %0, %1, off"
               :: "v"(lds_byte_off), "v"((uint64_t)(uintptr_t)gptr)
               : "memory");
}
__device__ __forceinline__ void wait_async0() {
  asm volatile("s_wait_asynccnt 0" ::: "memory");
}

// ---------------------------------------------------------------------------
// Kernel 1: fp32 -> bf16 conversion (4 elements / thread)
// ---------------------------------------------------------------------------
__global__ __launch_bounds__(256) void cvt_f32_to_bf16(const float* __restrict__ in,
                                                       bf16* __restrict__ out, int n4) {
  int i = blockIdx.x * 256 + threadIdx.x;
  if (i < n4) {
    float4 v = ((const float4*)in)[i];
    v4bf o;
    o[0] = f2bf(v.x); o[1] = f2bf(v.y); o[2] = f2bf(v.z); o[3] = f2bf(v.w);
    ((v4bf*)out)[i] = o;
  }
}

// ---------------------------------------------------------------------------
// Kernel 2/5: bf16 WMMA GEMM   out[M,N] = A[M,K] * Bm[N,K]^T + bias
// Block 256 thr (8 waves, 4x2), block tile 128x128, wave tile 32x64, K-step 32.
// A/B tiles staged in LDS with GLOBAL_LOAD_ASYNC_TO_LDS_B128, double-buffered;
// next stage's DMA overlaps current stage's WMMAs; fenced by ASYNCcnt+barrier.
// ---------------------------------------------------------------------------
#define LDSTRIDE 40   // bf16 elems per LDS row (32 data + 8 pad -> conflict-free)

template <bool OUT_BF16>
__global__ __launch_bounds__(256) void gemm_bf16_wmma(const bf16* __restrict__ A,
                                                      const bf16* __restrict__ Bm,
                                                      const float* __restrict__ bias,
                                                      void* __restrict__ outp,
                                                      int M, int N, int K) {
  __shared__ bf16 sA[2][128 * LDSTRIDE];
  __shared__ bf16 sB[2][128 * LDSTRIDE];

  const int tid  = threadIdx.x;
  const int lane = tid & 31;
  const int wave = tid >> 5;
  const int wm = wave & 3;              // 4 waves along M
  const int wn = wave >> 2;             // 2 waves along N
  const int tileM = blockIdx.y * 128;
  const int tileN = blockIdx.x * 128;
  const int lr = lane & 15;
  const int lh = (lane >> 4) * 8;       // K-interleave base per lane half

  // Copy geometry: 128x32 bf16 tile = 512 x 16B chunks; 256 threads x 2 chunks.
  // chunk c: row = c>>2, elem offset q = (c&3)*8
  const int c0 = tid,        r0 = c0 >> 2, q0 = (c0 & 3) * 8;
  const int c1 = tid + 256,  r1 = c1 >> 2, q1 = (c1 & 3) * 8;

  auto stage = [&](int buf, int k) {
    async_copy_b128(lds_off(&sA[buf][r0 * LDSTRIDE + q0]),
                    A + (size_t)(tileM + r0) * K + k + q0);
    async_copy_b128(lds_off(&sA[buf][r1 * LDSTRIDE + q1]),
                    A + (size_t)(tileM + r1) * K + k + q1);
    async_copy_b128(lds_off(&sB[buf][r0 * LDSTRIDE + q0]),
                    Bm + (size_t)(tileN + r0) * K + k + q0);
    async_copy_b128(lds_off(&sB[buf][r1 * LDSTRIDE + q1]),
                    Bm + (size_t)(tileN + r1) * K + k + q1);
  };

  v8f acc[2][4] = {};

  const int KT = K >> 5;
  stage(0, 0);
  wait_async0();
  __syncthreads();

  for (int kt = 0; kt < KT; ++kt) {
    const int cur = kt & 1;
    if (kt + 1 < KT) stage(1 - cur, (kt + 1) << 5);   // overlap DMA with WMMA

    v16bf af[2], bfm[4];
#pragma unroll
    for (int mt = 0; mt < 2; ++mt)
      af[mt] = load_frag16(&sA[cur][(wm * 32 + mt * 16 + lr) * LDSTRIDE + lh]);
#pragma unroll
    for (int nt = 0; nt < 4; ++nt)
      bfm[nt] = load_frag16(&sB[cur][(wn * 64 + nt * 16 + lr) * LDSTRIDE + lh]);
#pragma unroll
    for (int mt = 0; mt < 2; ++mt)
#pragma unroll
      for (int nt = 0; nt < 4; ++nt)
        acc[mt][nt] = WMMA_BF16(af[mt], bfm[nt], acc[mt][nt]);

    wait_async0();       // this wave's DMA into the next buffer has landed
    __syncthreads();     // everyone's DMA landed + everyone done reading cur
  }

  // C/D layout: VGPR r <-> row r + 8*(lane>=16); col = tile + (lane&15)
  const int rowoff = (lane >> 4) * 8;
#pragma unroll
  for (int mt = 0; mt < 2; ++mt) {
#pragma unroll
    for (int nt = 0; nt < 4; ++nt) {
      const int n = tileN + wn * 64 + nt * 16 + lr;
      const float bv = bias[n];
#pragma unroll
      for (int r = 0; r < 8; ++r) {
        const int m = tileM + wm * 32 + mt * 16 + rowoff + r;
        const float val = acc[mt][nt][r] + bv;
        if (OUT_BF16) ((bf16*)outp)[(size_t)m * N + n] = f2bf(val);
        else          ((float*)outp)[(size_t)m * N + n] = val;
      }
    }
  }
}

// ---------------------------------------------------------------------------
// Kernel 3: build V^T[b,h,hd,t] from qkv (v is cols 2048..3071)
// ---------------------------------------------------------------------------
__global__ __launch_bounds__(256) void transpose_v(const bf16* __restrict__ qkv,
                                                   bf16* __restrict__ vT) {
  int idx = blockIdx.x * 256 + threadIdx.x;    // output-linear: ((b*16+h)*64+hd)*T + t
  int t   = idx & (TT - 1);
  int rest = idx >> 11;
  int hd  = rest & (HD - 1);
  int bh  = rest >> 6;
  int h = bh & (NHEAD - 1), b = bh >> 4;
  vT[idx] = qkv[(size_t)(b * TT + t) * C3 + 2 * CC + h * HD + hd];
}

// ---------------------------------------------------------------------------
// Kernel 4: causal flash attention (bf16 WMMA, f32 online softmax)
// Block 128 thr = 4 waves; each wave owns a 16-row Q tile; 32-key tiles.
// ---------------------------------------------------------------------------
__global__ __launch_bounds__(128) void attn_flash(const bf16* __restrict__ qkv,
                                                  const bf16* __restrict__ vT,
                                                  bf16* __restrict__ yout) {
  const int lane = threadIdx.x & 31;
  const int wave = threadIdx.x >> 5;
  const int qt = blockIdx.x & 31;               // 32 q-tiles of 64 rows
  const int bh = blockIdx.x >> 5;
  const int h = bh & (NHEAD - 1), b = bh >> 4;
  const int qbase = qt * 64 + wave * 16;

  const bf16* qbaseptr = qkv + (size_t)b * TT * C3 + h * HD;       // + t*C3
  const bf16* kbaseptr = qbaseptr + CC;
  const bf16* vtp = vT + (size_t)(b * NHEAD + h) * HD * TT;        // + hd*T + t

  __shared__ bf16 ldsP[4][16 * 40];   // padded stride 40 bf16 -> conflict-free
  bf16* myP = ldsP[wave];

  const int lr = lane & 15;
  const int lh = (lane >> 4) * 8;
  const int rowoff = (lane >> 4) * 8; // C-layout row base for this lane half

  // Q fragments (A-matrix 16x64, two K-steps of 32)
  const bf16* qrow = qbaseptr + (size_t)(qbase + lr) * C3;
  v16bf qf0 = load_frag16(qrow + lh);
  v16bf qf1 = load_frag16(qrow + 32 + lh);

  v8f o[4] = {};
  float mstat[8], lstat[8], cfarr[8];
#pragma unroll
  for (int r = 0; r < 8; ++r) { mstat[r] = -__builtin_inff(); lstat[r] = 0.f; }

  const int ntiles = (qbase + 16 + 31) >> 5;    // 32-key tiles covering 0..qbase+15
  for (int j = 0; j < ntiles; ++j) {
    const int kb = j * 32;

    // K fragments (B-matrix): column n = key index
    const bf16* kr0 = kbaseptr + (size_t)(kb + lr) * C3 + lh;
    const bf16* kr1 = kbaseptr + (size_t)(kb + 16 + lr) * C3 + lh;
    v8f s0 = {}, s1 = {};
    s0 = WMMA_BF16(qf0, load_frag16(kr0),      s0);
    s0 = WMMA_BF16(qf1, load_frag16(kr0 + 32), s0);
    s1 = WMMA_BF16(qf0, load_frag16(kr1),      s1);
    s1 = WMMA_BF16(qf1, load_frag16(kr1 + 32), s1);

    // scale + causal mask; then online softmax update
    float pr0[8], pr1[8];
#pragma unroll
    for (int r = 0; r < 8; ++r) {
      const int row = qbase + rowoff + r;
      float v0 = s0[r] * 0.125f;                   // 1/sqrt(64)
      float v1 = s1[r] * 0.125f;
      if (kb + lr > row)      v0 = -__builtin_inff();
      if (kb + 16 + lr > row) v1 = -__builtin_inff();

      float mx = fmaxf(v0, v1);
#pragma unroll
      for (int d = 1; d < 16; d <<= 1) mx = fmaxf(mx, __shfl_xor(mx, d, 32));
      const float mnew = fmaxf(mstat[r], mx);

      pr0[r] = __expf(v0 - mnew);
      pr1[r] = __expf(v1 - mnew);
      float sum = pr0[r] + pr1[r];
#pragma unroll
      for (int d = 1; d < 16; d <<= 1) sum += __shfl_xor(sum, d, 32);

      const float cf = __expf(mstat[r] - mnew);
      lstat[r] = lstat[r] * cf + sum;
      mstat[r] = mnew;
      cfarr[r] = cf;
    }

    // stage P (16x32 bf16) row-major in LDS, reload in A-fragment layout
#pragma unroll
    for (int r = 0; r < 8; ++r) {
      myP[(rowoff + r) * 40 + lr]      = f2bf(pr0[r]);
      myP[(rowoff + r) * 40 + 16 + lr] = f2bf(pr1[r]);
    }
    asm volatile("s_wait_dscnt 0" ::: "memory");   // in-wave LDS store->load hazard
    v16bf pf = load_frag16(&myP[lr * 40 + lh]);

    // V fragments (B-matrix 32x64): column n = hd, contiguous keys from V^T
#pragma unroll
    for (int nt = 0; nt < 4; ++nt) {
      v16bf vf = load_frag16(vtp + (size_t)(nt * 16 + lr) * TT + kb + lh);
      v8f cacc;
#pragma unroll
      for (int e = 0; e < 8; ++e) cacc[e] = o[nt][e] * cfarr[e];
      o[nt] = WMMA_BF16(pf, vf, cacc);
    }
  }

  // epilogue: normalize, store y[b, t, h*64 + hd] as bf16
#pragma unroll
  for (int nt = 0; nt < 4; ++nt) {
    const int col = h * HD + nt * 16 + lr;
#pragma unroll
    for (int r = 0; r < 8; ++r) {
      const int row = qbase + rowoff + r;
      yout[(size_t)(b * TT + row) * CC + col] = f2bf(o[nt][r] / lstat[r]);
    }
  }
}

// ---------------------------------------------------------------------------
// Host-side launch
// ---------------------------------------------------------------------------
extern "C" void kernel_launch(void* const* d_in, const int* in_sizes, int n_in,
                              void* d_out, int out_size, void* d_ws, size_t ws_size,
                              hipStream_t stream) {
  const float* x      = (const float*)d_in[0];   // [2,2048,1024]
  const float* w_attn = (const float*)d_in[1];   // [3072,1024]
  const float* b_attn = (const float*)d_in[2];   // [3072]
  const float* w_proj = (const float*)d_in[3];   // [1024,1024]
  const float* b_proj = (const float*)d_in[4];   // [1024]

  const int nX  = BB * TT * CC;       // 4194304
  const int nWA = 3 * CC * CC;        // 3145728
  const int nWP = CC * CC;            // 1048576
  const int nQKV = BB * TT * C3;      // 12582912

  bf16* x_bf   = (bf16*)d_ws;
  bf16* wa_bf  = x_bf   + nX;
  bf16* wp_bf  = wa_bf  + nWA;
  bf16* qkv_bf = wp_bf  + nWP;
  bf16* vT_bf  = qkv_bf + nQKV;
  bf16* y_bf   = vT_bf  + nX;

  // 1) fp32 -> bf16
  cvt_f32_to_bf16<<<nX  / 1024, 256, 0, stream>>>(x,      x_bf,  nX  / 4);
  cvt_f32_to_bf16<<<nWA / 1024, 256, 0, stream>>>(w_attn, wa_bf, nWA / 4);
  cvt_f32_to_bf16<<<nWP / 1024, 256, 0, stream>>>(w_proj, wp_bf, nWP / 4);

  // 2) QKV GEMM: [4096,3072] = x[4096,1024] @ w_attn^T + b_attn  (bf16 out)
  gemm_bf16_wmma<true><<<dim3(C3 / 128, (BB * TT) / 128), 256, 0, stream>>>(
      x_bf, wa_bf, b_attn, (void*)qkv_bf, BB * TT, C3, CC);

  // 3) V^T per head
  transpose_v<<<nX / 256, 256, 0, stream>>>(qkv_bf, vT_bf);

  // 4) causal flash attention -> y_bf [4096,1024]
  attn_flash<<<BB * NHEAD * (TT / 64), 128, 0, stream>>>(qkv_bf, vT_bf, y_bf);

  // 5) projection: d_out[4096,1024] = y @ w_proj^T + b_proj  (fp32 out)
  gemm_bf16_wmma<false><<<dim3(CC / 128, (BB * TT) / 128), 256, 0, stream>>>(
      y_bf, wp_bf, b_proj, d_out, BB * TT, CC, CC);
}
